// DogStateEncoder_29798483100401
// MI455X (gfx1250) — matrix operand order repaired and checked by reading, compiled, not run on hardware
//
#include <hip/hip_runtime.h>

typedef __attribute__((ext_vector_type(16))) _Float16 v16h;
typedef __attribute__((ext_vector_type(8)))  _Float16 v8h;
typedef __attribute__((ext_vector_type(8)))  float    v8f;

// ---------------- LDS layout ----------------
// f16 weight region (element offsets in halves)
constexpr int OFF_WHH0 = 0;                        // [192][64]
constexpr int OFF_WIH1 = OFF_WHH0 + 192 * 64;      // [192][64]
constexpr int OFF_WHH1 = OFF_WIH1 + 192 * 64;      // [192][64]
constexpr int OFF_FW1  = OFF_WHH1 + 192 * 64;      // [128][128]
constexpr int OFF_FW2  = OFF_FW1 + 128 * 128;      // [64][128]
constexpr int OFF_VW2  = OFF_FW2 + 64 * 128;       // [32][32]
constexpr int OFF_GW2  = OFF_VW2 + 32 * 32;        // [32][32]
constexpr int HALVES_TOTAL = OFF_GW2 + 32 * 32;    // 63488 halves = 126976 B

// f32 constant region (element offsets in floats), starts at HALVES_TOTAL*2
constexpr int FOFF_WIH0 = 0;     // [192][2]
constexpr int FOFF_BIH0 = 384;   // [192]
constexpr int FOFF_BHH0 = 576;
constexpr int FOFF_BIH1 = 768;
constexpr int FOFF_BHH1 = 960;
constexpr int FOFF_VW1  = 1152;  // [32][2]
constexpr int FOFF_VB1  = 1216;
constexpr int FOFF_VB2  = 1248;
constexpr int FOFF_GW1  = 1280;
constexpr int FOFF_GB1  = 1344;
constexpr int FOFF_GB2  = 1376;
constexpr int FOFF_FB1  = 1408;  // [128]
constexpr int FOFF_FB2  = 1536;  // [64]
constexpr int FOFF_LNG  = 1600;  // [64]
constexpr int FOFF_LNB  = 1664;  // [64]
constexpr int F32_TOTAL = 1728;

constexpr int WEIGHT_BYTES = HALVES_TOTAL * 2 + F32_TOTAL * 4;  // 133888

// per-wave staging region (byte offsets inside the wave block)
constexpr int PWO_H0   = 0;      // [16][64] f16  = 2048 B
constexpr int PWO_H1   = 2048;   // [16][64] f16
constexpr int PWO_COMB = 4096;   // [16][128] f16 = 4096 B (reused by fusion L1 out)
constexpr int PWO_VG   = 8192;   // [16][32] f16  = 1024 B
constexpr int PWO_X    = 9216;   // [T=8][16][2] f32 = 1024 B (transposed!)
constexpr int PWO_IN   = 10240;  // [16][4] f32   = 256 B
constexpr int PW_BYTES = 10496;

constexpr int WAVES = 16;        // 512 threads -> 4 waves/SIMD (1 WG per WGP)
constexpr int NTHREADS = WAVES * 32;
constexpr int SMEM_BYTES = WEIGHT_BYTES + WAVES * PW_BYTES;  // 301824 B <= 320KB

union V16H { v16h v; v8h h[2]; };

__device__ __forceinline__ v8f splat8(float x) {
  v8f r;
#pragma unroll
  for (int i = 0; i < 8; ++i) r[i] = x;
  return r;
}

// A operand (16xK activation tile, row-major f16 [16][Kfull]) for K-chunk kc.
// ISA 16-bit A 16x32 layout: lane L -> m=L&15, two contiguous 8-half runs.
__device__ __forceinline__ v16h ldsA(const _Float16* Ast, int Kfull, int kc, int lane) {
  int m  = lane & 15;
  int kb = ((lane & 16) >> 1) + kc * 32;  // lanes>=16 start at K+8
  const _Float16* row = Ast + m * Kfull + kb;
  V16H u;
  u.h[0] = *(const v8h*)(row);        // K = kb .. kb+7
  u.h[1] = *(const v8h*)(row + 16);   // K = kb+16 .. kb+23
  return u.v;
}

// B operand (KxN weight tile) from W stored row-major [N][Kfull] f16.
// B layout: lane L -> n=L&15, two 8-half runs at K = 8*(L>>4) and 16 + 8*(L>>4) (+ 32*kc).
__device__ __forceinline__ v16h ldsB(const _Float16* Wst, int Kfull, int kc, int jn, int lane) {
  int n = lane & 15;
  int g = lane >> 4;
  const _Float16* p = Wst + (jn * 16 + n) * Kfull + kc * 32 + g * 8;
  V16H u;
  u.h[0] = *(const v8h*)(p);
  u.h[1] = *(const v8h*)(p + 16);
  return u.v;
}

#define WMMA_F16(A, Bm, C) \
  __builtin_amdgcn_wmma_f32_16x16x32_f16(false, (A), false, (Bm), (short)0, (C), false, false)

// hardware-native transcendentals: v_rcp_f32 / v_rsq_f32 (~1 ulp).
__device__ __forceinline__ float frcp(float x) { return __builtin_amdgcn_rcpf(x); }
__device__ __forceinline__ float sigm(float x) { return frcp(1.0f + __expf(-x)); }
// branch-free tanh: 1 - 2/(e^{2x}+1); saturates to +/-1 for large |x|.
__device__ __forceinline__ float ftanh(float x) {
  return 1.0f - 2.0f * frcp(__expf(2.0f * x) + 1.0f);
}

__global__ __attribute__((amdgpu_waves_per_eu(4))) __launch_bounds__(NTHREADS)
void dog_encoder_kernel(
    const float* __restrict__ past, const float* __restrict__ vel,
    const float* __restrict__ cpos, const float* __restrict__ gpos,
    const float* __restrict__ w_ih0, const float* __restrict__ w_hh0,
    const float* __restrict__ b_ih0, const float* __restrict__ b_hh0,
    const float* __restrict__ w_ih1, const float* __restrict__ w_hh1,
    const float* __restrict__ b_ih1, const float* __restrict__ b_hh1,
    const float* __restrict__ vw1, const float* __restrict__ vb1,
    const float* __restrict__ vw2, const float* __restrict__ vb2,
    const float* __restrict__ gw1, const float* __restrict__ gb1,
    const float* __restrict__ gw2, const float* __restrict__ gb2,
    const float* __restrict__ fw1, const float* __restrict__ fb1,
    const float* __restrict__ fw2, const float* __restrict__ fb2,
    const float* __restrict__ ln_g, const float* __restrict__ ln_b,
    float* __restrict__ out)
{
  extern __shared__ char smem[];
  _Float16* hw = (_Float16*)smem;
  float*    fc = (float*)(smem + HALVES_TOTAL * 2);

  const int tid = threadIdx.x;

  // ---- cooperative weight staging: f32 -> f16 (row-major [N][K]) ----
  {
    // vectorized converter: nel must be a multiple of 4 (all are)
    auto cvt4 = [&](const float* g, _Float16* s, int nel) {
      for (int i = tid * 4; i < nel; i += NTHREADS * 4) {
        const float4 v = *(const float4*)(g + i);
        union { _Float16 h[4]; uint2 u; } p;
        p.h[0] = (_Float16)v.x; p.h[1] = (_Float16)v.y;
        p.h[2] = (_Float16)v.z; p.h[3] = (_Float16)v.w;
        *(uint2*)(s + i) = p.u;
      }
    };
    auto cpy = [&](const float* g, float* s, int nel) {
      for (int i = tid; i < nel; i += NTHREADS) s[i] = g[i];
    };
    cvt4(w_hh0, hw + OFF_WHH0, 192 * 64);
    cvt4(w_ih1, hw + OFF_WIH1, 192 * 64);
    cvt4(w_hh1, hw + OFF_WHH1, 192 * 64);
    cvt4(fw1,   hw + OFF_FW1,  128 * 128);
    cvt4(fw2,   hw + OFF_FW2,  64 * 128);
    cvt4(vw2,   hw + OFF_VW2,  32 * 32);
    cvt4(gw2,   hw + OFF_GW2,  32 * 32);
    cpy(w_ih0, fc + FOFF_WIH0, 384);
    cpy(b_ih0, fc + FOFF_BIH0, 192);
    cpy(b_hh0, fc + FOFF_BHH0, 192);
    cpy(b_ih1, fc + FOFF_BIH1, 192);
    cpy(b_hh1, fc + FOFF_BHH1, 192);
    cpy(vw1,   fc + FOFF_VW1,  64);
    cpy(vb1,   fc + FOFF_VB1,  32);
    cpy(vb2,   fc + FOFF_VB2,  32);
    cpy(gw1,   fc + FOFF_GW1,  64);
    cpy(gb1,   fc + FOFF_GB1,  32);
    cpy(gb2,   fc + FOFF_GB2,  32);
    cpy(fb1,   fc + FOFF_FB1,  128);
    cpy(fb2,   fc + FOFF_FB2,  64);
    cpy(ln_g,  fc + FOFF_LNG,  64);
    cpy(ln_b,  fc + FOFF_LNB,  64);
  }
  __syncthreads();

  const _Float16* WhhSt0 = hw + OFF_WHH0;
  const _Float16* WihSt1 = hw + OFF_WIH1;
  const _Float16* WhhSt1 = hw + OFF_WHH1;
  const _Float16* Fw1St  = hw + OFF_FW1;
  const _Float16* Fw2St  = hw + OFF_FW2;
  const _Float16* Vw2St  = hw + OFF_VW2;
  const _Float16* Gw2St  = hw + OFF_GW2;
  const float* wih0f = fc + FOFF_WIH0;
  const float* bih0f = fc + FOFF_BIH0;
  const float* bhh0f = fc + FOFF_BHH0;
  const float* bih1f = fc + FOFF_BIH1;
  const float* bhh1f = fc + FOFF_BHH1;
  const float* vw1f  = fc + FOFF_VW1;
  const float* vb1f  = fc + FOFF_VB1;
  const float* vb2f  = fc + FOFF_VB2;
  const float* gw1f  = fc + FOFF_GW1;
  const float* gb1f  = fc + FOFF_GB1;
  const float* gb2f  = fc + FOFF_GB2;
  const float* fb1f  = fc + FOFF_FB1;
  const float* fb2f  = fc + FOFF_FB2;
  const float* lngf  = fc + FOFF_LNG;
  const float* lnbf  = fc + FOFF_LNB;

  const int wave = tid >> 5;
  const int lane = tid & 31;
  char* wbase = smem + WEIGHT_BYTES + wave * PW_BYTES;
  _Float16* h0st   = (_Float16*)(wbase + PWO_H0);
  _Float16* h1st   = (_Float16*)(wbase + PWO_H1);
  _Float16* combSt = (_Float16*)(wbase + PWO_COMB);
  _Float16* vgSt   = (_Float16*)(wbase + PWO_VG);
  float*    xSt    = (float*)(wbase + PWO_X);   // [T][16][2]
  float*    inSt   = (float*)(wbase + PWO_IN);

  const int r0 = (blockIdx.x * WAVES + wave) * 16;

  // zero hidden-state staging (h0 = h1 = 0)
  for (int i = lane; i < 512; i += 32) {
    ((unsigned*)h0st)[i] = 0u;
    ((unsigned*)h1st)[i] = 0u;
  }

  // stage per-row inputs: rel_traj transposed [t][m][2], and (vx,vy,relgx,relgy)
  if (lane < 16) {
    const int m = lane;
    const int row = r0 + m;
    const float cx = cpos[row * 2 + 0];
    const float cy = cpos[row * 2 + 1];
#pragma unroll
    for (int t = 0; t < 8; ++t) {
      xSt[t * 32 + m * 2 + 0] = past[(row * 8 + t) * 2 + 0] - cx;
      xSt[t * 32 + m * 2 + 1] = past[(row * 8 + t) * 2 + 1] - cy;
    }
    inSt[m * 4 + 0] = vel[row * 2 + 0];
    inSt[m * 4 + 1] = vel[row * 2 + 1];
    inSt[m * 4 + 2] = gpos[row * 2 + 0] - cx;
    inSt[m * 4 + 3] = gpos[row * 2 + 1] - cy;
  }

  const int n  = lane & 15;        // N index inside 16x16 C tile
  const int mg = (lane >> 4) * 8;  // M base: lanes 0-15 -> rows 0-7; 16-31 -> 8-15

  v8f h0c[4], h1c[4];
#pragma unroll
  for (int j = 0; j < 4; ++j) { h0c[j] = splat8(0.f); h1c[j] = splat8(0.f); }

  // layer-0 A operand; rotated from layer-1's p0/p1 loads after step 0
  v16h a0 = ldsA(h0st, 64, 0, lane);
  v16h a1 = ldsA(h0st, 64, 1, lane);

#pragma unroll 1
  for (int t = 0; t < 8; ++t) {
    // ================= GRU layer 0 =================
    {
      // this timestep's (x,y) for rows mg..mg+7: 16 contiguous floats
      v8f xv, yv;
      {
        const float* xp = xSt + t * 32 + mg * 2;
        const float4 A4 = *(const float4*)(xp + 0);
        const float4 B4 = *(const float4*)(xp + 4);
        const float4 C4 = *(const float4*)(xp + 8);
        const float4 D4 = *(const float4*)(xp + 12);
        xv[0] = A4.x; yv[0] = A4.y; xv[1] = A4.z; yv[1] = A4.w;
        xv[2] = B4.x; yv[2] = B4.y; xv[3] = B4.z; yv[3] = B4.w;
        xv[4] = C4.x; yv[4] = C4.y; xv[5] = C4.z; yv[5] = C4.w;
        xv[6] = D4.x; yv[6] = D4.y; xv[7] = D4.z; yv[7] = D4.w;
      }
#pragma unroll
      for (int j = 0; j < 4; ++j) {
        const int colr = j * 16 + n;
        const int colz = (j + 4) * 16 + n;
        const int coln = (j + 8) * 16 + n;
        const float wrx = wih0f[colr * 2], wry = wih0f[colr * 2 + 1];
        const float wzx = wih0f[colz * 2], wzy = wih0f[colz * 2 + 1];
        const float wnx = wih0f[coln * 2], wny = wih0f[coln * 2 + 1];
        const float br = bih0f[colr] + bhh0f[colr];
        const float bz = bih0f[colz] + bhh0f[colz];
        const float bn = bih0f[coln];
        v8f cr, cz, xnv;
#pragma unroll
        for (int vg = 0; vg < 8; ++vg) {
          cr[vg]  = fmaf(xv[vg], wrx, fmaf(yv[vg], wry, br));
          cz[vg]  = fmaf(xv[vg], wzx, fmaf(yv[vg], wzy, bz));
          xnv[vg] = fmaf(xv[vg], wnx, fmaf(yv[vg], wny, bn));
        }
        cr = WMMA_F16(a0, ldsB(WhhSt0, 64, 0, j, lane), cr);
        cr = WMMA_F16(a1, ldsB(WhhSt0, 64, 1, j, lane), cr);
        cz = WMMA_F16(a0, ldsB(WhhSt0, 64, 0, j + 4, lane), cz);
        cz = WMMA_F16(a1, ldsB(WhhSt0, 64, 1, j + 4, lane), cz);
        // shorten live ranges: convert gate pre-activations immediately
#pragma unroll
        for (int vg = 0; vg < 8; ++vg) {
          cr[vg] = sigm(cr[vg]);
          cz[vg] = sigm(cz[vg]);
        }
        v8f ch = splat8(bhh0f[coln]);
        ch = WMMA_F16(a0, ldsB(WhhSt0, 64, 0, j + 8, lane), ch);
        ch = WMMA_F16(a1, ldsB(WhhSt0, 64, 1, j + 8, lane), ch);
#pragma unroll
        for (int vg = 0; vg < 8; ++vg) {
          const float nn = ftanh(xnv[vg] + cr[vg] * ch[vg]);
          h0c[j][vg] = (1.f - cz[vg]) * nn + cz[vg] * h0c[j][vg];
          h0st[(mg + vg) * 64 + j * 16 + n] = (_Float16)h0c[j][vg];
        }
      }
    }
    // ================= GRU layer 1 (input = new h0) =================
    {
      v16h p0 = ldsA(h0st, 64, 0, lane);
      v16h p1 = ldsA(h0st, 64, 1, lane);
      v16h q0 = ldsA(h1st, 64, 0, lane);
      v16h q1 = ldsA(h1st, 64, 1, lane);
#pragma unroll
      for (int j = 0; j < 4; ++j) {
        const int colr = j * 16 + n;
        const int colz = (j + 4) * 16 + n;
        const int coln = (j + 8) * 16 + n;
        v8f cr = splat8(bih1f[colr] + bhh1f[colr]);
        cr = WMMA_F16(p0, ldsB(WihSt1, 64, 0, j, lane), cr);
        cr = WMMA_F16(p1, ldsB(WihSt1, 64, 1, j, lane), cr);
        cr = WMMA_F16(q0, ldsB(WhhSt1, 64, 0, j, lane), cr);
        cr = WMMA_F16(q1, ldsB(WhhSt1, 64, 1, j, lane), cr);
        v8f cz = splat8(bih1f[colz] + bhh1f[colz]);
        cz = WMMA_F16(p0, ldsB(WihSt1, 64, 0, j + 4, lane), cz);
        cz = WMMA_F16(p1, ldsB(WihSt1, 64, 1, j + 4, lane), cz);
        cz = WMMA_F16(q0, ldsB(WhhSt1, 64, 0, j + 4, lane), cz);
        cz = WMMA_F16(q1, ldsB(WhhSt1, 64, 1, j + 4, lane), cz);
#pragma unroll
        for (int vg = 0; vg < 8; ++vg) {
          cr[vg] = sigm(cr[vg]);
          cz[vg] = sigm(cz[vg]);
        }
        v8f cx = splat8(bih1f[coln]);
        cx = WMMA_F16(p0, ldsB(WihSt1, 64, 0, j + 8, lane), cx);
        cx = WMMA_F16(p1, ldsB(WihSt1, 64, 1, j + 8, lane), cx);
        v8f ch = splat8(bhh1f[coln]);
        ch = WMMA_F16(q0, ldsB(WhhSt1, 64, 0, j + 8, lane), ch);
        ch = WMMA_F16(q1, ldsB(WhhSt1, 64, 1, j + 8, lane), ch);
#pragma unroll
        for (int vg = 0; vg < 8; ++vg) {
          const float nn = ftanh(cx[vg] + cr[vg] * ch[vg]);
          h1c[j][vg] = (1.f - cz[vg]) * nn + cz[vg] * h1c[j][vg];
          h1st[(mg + vg) * 64 + j * 16 + n] = (_Float16)h1c[j][vg];
        }
      }
      // rotate: next step's layer-0 A operand is the h0 state we just loaded
      a0 = p0;
      a1 = p1;
    }
  }

  // ---- combined[:, 0:64] = traj_feature (final h1) ----
#pragma unroll
  for (int j = 0; j < 4; ++j)
#pragma unroll
    for (int vg = 0; vg < 8; ++vg)
      combSt[(mg + vg) * 128 + j * 16 + n] = (_Float16)h1c[j][vg];

  // ---- velocity encoder: 2 -> 32 (VALU) -> 32 (WMMA) ----
  {
#pragma unroll
    for (int jn = 0; jn < 2; ++jn) {
      const int col = jn * 16 + n;
      const float wx = vw1f[col * 2 + 0], wy = vw1f[col * 2 + 1], bb = vb1f[col];
#pragma unroll
      for (int vg = 0; vg < 8; ++vg) {
        const int m = mg + vg;
        const float v = fmaf(inSt[m * 4 + 0], wx, fmaf(inSt[m * 4 + 1], wy, bb));
        vgSt[m * 32 + col] = (_Float16)fmaxf(v, 0.f);
      }
    }
    v16h av = ldsA(vgSt, 32, 0, lane);
#pragma unroll
    for (int jn = 0; jn < 2; ++jn) {
      const int col = jn * 16 + n;
      v8f c = splat8(vb2f[col]);
      c = WMMA_F16(av, ldsB(Vw2St, 32, 0, jn, lane), c);
#pragma unroll
      for (int vg = 0; vg < 8; ++vg)
        combSt[(mg + vg) * 128 + 64 + col] = (_Float16)fmaxf(c[vg], 0.f);
    }
  }
  // ---- goal encoder: 2 -> 32 (VALU) -> 32 (WMMA) ----
  {
#pragma unroll
    for (int jn = 0; jn < 2; ++jn) {
      const int col = jn * 16 + n;
      const float wx = gw1f[col * 2 + 0], wy = gw1f[col * 2 + 1], bb = gb1f[col];
#pragma unroll
      for (int vg = 0; vg < 8; ++vg) {
        const int m = mg + vg;
        const float v = fmaf(inSt[m * 4 + 2], wx, fmaf(inSt[m * 4 + 3], wy, bb));
        vgSt[m * 32 + col] = (_Float16)fmaxf(v, 0.f);
      }
    }
    v16h ag = ldsA(vgSt, 32, 0, lane);
#pragma unroll
    for (int jn = 0; jn < 2; ++jn) {
      const int col = jn * 16 + n;
      v8f c = splat8(gb2f[col]);
      c = WMMA_F16(ag, ldsB(Gw2St, 32, 0, jn, lane), c);
#pragma unroll
      for (int vg = 0; vg < 8; ++vg)
        combSt[(mg + vg) * 128 + 96 + col] = (_Float16)fmaxf(c[vg], 0.f);
    }
  }

  // ---- fusion layer 1: [16,128] x [128,128]; output overwrites combSt ----
  v16h af[4];
#pragma unroll
  for (int kc = 0; kc < 4; ++kc) af[kc] = ldsA(combSt, 128, kc, lane);
#pragma unroll
  for (int jn = 0; jn < 8; ++jn) {
    const int col = jn * 16 + n;
    v8f c = splat8(fb1f[col]);
#pragma unroll
    for (int kc = 0; kc < 4; ++kc)
      c = WMMA_F16(af[kc], ldsB(Fw1St, 128, kc, jn, lane), c);
#pragma unroll
    for (int vg = 0; vg < 8; ++vg)
      combSt[(mg + vg) * 128 + col] = (_Float16)fmaxf(c[vg], 0.f);
  }
  // ---- fusion layer 2: [16,128] x [128,64] ----
  v8f fused[4];
#pragma unroll
  for (int kc = 0; kc < 4; ++kc) af[kc] = ldsA(combSt, 128, kc, lane);
#pragma unroll
  for (int jn = 0; jn < 4; ++jn) {
    v8f c = splat8(fb2f[jn * 16 + n]);
#pragma unroll
    for (int kc = 0; kc < 4; ++kc)
      c = WMMA_F16(af[kc], ldsB(Fw2St, 128, kc, jn, lane), c);
#pragma unroll
    for (int vg = 0; vg < 8; ++vg)
      fused[jn][vg] = fmaxf(c[vg], 0.f);
  }

  // ---- LayerNorm over 64 cols (16 lanes x 4 tiles per row) + store ----
#pragma unroll
  for (int vg = 0; vg < 8; ++vg) {
    float s = fused[0][vg] + fused[1][vg] + fused[2][vg] + fused[3][vg];
    float q = fused[0][vg] * fused[0][vg] + fused[1][vg] * fused[1][vg] +
              fused[2][vg] * fused[2][vg] + fused[3][vg] * fused[3][vg];
#pragma unroll
    for (int msk = 1; msk <= 8; msk <<= 1) {
      s += __shfl_xor(s, msk, 32);
      q += __shfl_xor(q, msk, 32);
    }
    const float mu  = s * (1.f / 64.f);
    const float var = q * (1.f / 64.f) - mu * mu;
    const float inv = __builtin_amdgcn_rsqf(var + 1e-5f);
    const int row = r0 + mg + vg;
#pragma unroll
    for (int j = 0; j < 4; ++j) {
      const int col = j * 16 + n;
      out[row * 64 + col] = (fused[j][vg] - mu) * inv * lngf[col] + lnbf[col];
    }
  }
}

extern "C" void kernel_launch(void* const* d_in, const int* in_sizes, int n_in,
                              void* d_out, int out_size, void* d_ws, size_t ws_size,
                              hipStream_t stream) {
  (void)in_sizes; (void)n_in; (void)out_size; (void)d_ws; (void)ws_size;
  hipFuncSetAttribute((const void*)dog_encoder_kernel,
                      hipFuncAttributeMaxDynamicSharedMemorySize, SMEM_BYTES);
  const int B = 131072;
  dim3 grid(B / (16 * WAVES));  // 512 workgroups, 16 waves x 16 rows each
  dim3 block(NTHREADS);
  dog_encoder_kernel<<<grid, block, SMEM_BYTES, stream>>>(
      (const float*)d_in[0],  (const float*)d_in[1],  (const float*)d_in[2],
      (const float*)d_in[3],  (const float*)d_in[4],  (const float*)d_in[5],
      (const float*)d_in[6],  (const float*)d_in[7],  (const float*)d_in[8],
      (const float*)d_in[9],  (const float*)d_in[10], (const float*)d_in[11],
      (const float*)d_in[12], (const float*)d_in[13], (const float*)d_in[14],
      (const float*)d_in[15], (const float*)d_in[16], (const float*)d_in[17],
      (const float*)d_in[18], (const float*)d_in[19], (const float*)d_in[20],
      (const float*)d_in[21], (const float*)d_in[22], (const float*)d_in[23],
      (const float*)d_in[24], (const float*)d_in[25], (float*)d_out);
}
